// MultiHeadedAttentionSANM_33870112096572
// MI455X (gfx1250) — compile-verified, matrix operand order checked by reading, not run on hardware
//
#include <hip/hip_runtime.h>

// ---------------------------------------------------------------------------
// MI455X (gfx1250) MultiHeadedAttentionSANM
//   B=16, T=1024, FEAT=512, HEADS=4, DK=128, KERNEL=11
// All GEMMs use v_wmma_f32_16x16x32_bf16 (bf16 in, f32 accumulate).
// sched_barrier(0) separates fragment-load batches from WMMA bursts so 8
// WMMAs issue back-to-back behind a single s_wait_{dscnt,loadcnt}.
// ---------------------------------------------------------------------------

#define BATCH 16
#define SEQ   1024
#define HEADS 4
#define DK    128
#define FEAT  512
#define QKVF  1536
#define ROWS  (BATCH * SEQ)          // 16384
#define QSCALE 0.08838834764831845f  // 1/sqrt(128)

typedef __bf16 bf16_t;
typedef bf16_t v16bf __attribute__((ext_vector_type(16)));
typedef float  v8f   __attribute__((ext_vector_type(8)));

static __device__ __forceinline__ void sched_fence() {
#if __has_builtin(__builtin_amdgcn_sched_barrier)
  __builtin_amdgcn_sched_barrier(0);
#endif
}

static __device__ __forceinline__ unsigned short f2bf(float f) {
  union { float f; unsigned u; } c; c.f = f;
  unsigned r = (c.u + 0x7FFFu + ((c.u >> 16) & 1u)) >> 16;  // round-nearest-even
  return (unsigned short)r;
}

union FragU { v16bf v; uint4 q[2]; };

// Load one 16x32 A-fragment (or 32x16 B-fragment stored n-major) for
// v_wmma_*_16x16x32_bf16. Per ISA 7.12.2 (16-bit A 16x32 layout):
//   lane&15 selects row (A: M, B: N), lane>>4 selects the 8-wide K sub-block,
//   elements 0..7 -> K = kb..kb+7, elements 8..15 -> K = kb+16..kb+23.
// Rows are stored contiguously in K, so each fragment = two aligned 16B loads.
static __device__ __forceinline__ v16bf frag_at(const unsigned short* base, int row,
                                                int stride, int k0, int lane) {
  int r  = row + (lane & 15);
  int kb = (lane >> 4) << 3;
  const unsigned short* p = base + (size_t)r * stride + k0 + kb;
  FragU f;
  f.q[0] = *(const uint4*)(p);
  f.q[1] = *(const uint4*)(p + 16);
  return f.v;
}

static __device__ __forceinline__ v8f wmma_bf16(v16bf a, v16bf b, v8f c) {
  return __builtin_amdgcn_wmma_f32_16x16x32_bf16(false, a, false, b, (short)0, c,
                                                 false, false);
}

// Stage a 128x32 A-tile (f32 -> bf16) into LDS [m][k] stride 48.
// 4 fixed passes of float4 per thread: no guards, 16B global loads, 8B LDS stores.
static __device__ __forceinline__ void stage_a(unsigned short* aL,
                                               const float* __restrict__ src,
                                               int m0, int lda, int k0, int tid) {
#pragma unroll
  for (int p = 0; p < 4; ++p) {
    int i = (tid + p * 256) * 4;      // 0..4092, step 4
    int m = i >> 5, kk = i & 31;
    float4 v = *(const float4*)(src + (size_t)(m0 + m) * lda + k0 + kk);
    union { unsigned short s[4]; uint2 u; } pk;
    pk.s[0] = f2bf(v.x); pk.s[1] = f2bf(v.y);
    pk.s[2] = f2bf(v.z); pk.s[3] = f2bf(v.w);
    *(uint2*)(aL + m * 48 + kk) = pk.u;
  }
}

// Stage a 32x128 B-tile from row-major W[k][n] into LDS transposed [n][k].
static __device__ __forceinline__ void stage_b(unsigned short* bL,
                                               const float* __restrict__ src,
                                               int n0, int ldb, int k0, int tid) {
#pragma unroll
  for (int p = 0; p < 4; ++p) {
    int i = (tid + p * 256) * 4;      // kk = i>>7 (0..31), n = i&127 (mult of 4)
    int kk = i >> 7, n = i & 127;
    float4 v = *(const float4*)(src + (size_t)(k0 + kk) * ldb + n0 + n);
    bL[(n + 0) * 48 + kk] = f2bf(v.x);
    bL[(n + 1) * 48 + kk] = f2bf(v.y);
    bL[(n + 2) * 48 + kk] = f2bf(v.z);
    bL[(n + 3) * 48 + kk] = f2bf(v.w);
  }
}

// ---------------------------------------------------------------------------
// Kernel 1: qkv = x @ W_qkv + b_qkv.  Writes:
//   q (scaled) -> bf16 [b,h,t,d], k -> bf16 [b,h,t,d],
//   v -> bf16 transposed [b,h,d,t] (for P·V B-fragments) and f32 [b,t,c] (FSMN)
// ---------------------------------------------------------------------------
__global__ void __launch_bounds__(256) qkv_gemm_kernel(
    const float* __restrict__ x, const float* __restrict__ Wqkv,
    const float* __restrict__ bqkv,
    unsigned short* __restrict__ qbf, unsigned short* __restrict__ kbf,
    unsigned short* __restrict__ vbfT, float* __restrict__ vf32) {
  __shared__ unsigned short aL[128 * 48];  // [m][k], stride 48 halfwords (16B mult)
  __shared__ unsigned short bL[128 * 48];  // [n][k]
  const int m0 = blockIdx.x * 128;
  const int n0 = blockIdx.y * 128;
  const int tid = threadIdx.x;
  const int wave = tid >> 5, lane = tid & 31;
  const int hl = lane >> 4, ln = lane & 15;

  const v8f vzero = {0.f, 0.f, 0.f, 0.f, 0.f, 0.f, 0.f, 0.f};
  v8f acc[8];
#pragma unroll
  for (int j = 0; j < 8; ++j) acc[j] = vzero;

  for (int k0 = 0; k0 < FEAT; k0 += 32) {
    stage_a(aL, x, m0, FEAT, k0, tid);
    stage_b(bL, Wqkv, n0, QKVF, k0, tid);
    __syncthreads();
    v16bf af = frag_at(aL, wave * 16, 48, 0, lane);
    v16bf bfr[8];
#pragma unroll
    for (int j = 0; j < 8; ++j) bfr[j] = frag_at(bL, j * 16, 48, 0, lane);
    sched_fence();  // all 18 ds_load_b128 issued, one wait, then 8 WMMAs
#pragma unroll
    for (int j = 0; j < 8; ++j) acc[j] = wmma_bf16(af, bfr[j], acc[j]);
    sched_fence();
    __syncthreads();
  }

#pragma unroll
  for (int j = 0; j < 8; ++j) {
#pragma unroll
    for (int i = 0; i < 8; ++i) {
      int col = n0 + j * 16 + ln;                // 0..1535
      int row = m0 + wave * 16 + i + hl * 8;     // 0..16383
      float val = acc[j][i] + bqkv[col];
      int b = row >> 10, t = row & 1023;
      if (col < FEAT) {                          // Q (pre-scaled)
        int h = col >> 7, d = col & 127;
        qbf[((size_t)(b * HEADS + h) * SEQ + t) * DK + d] = f2bf(val * QSCALE);
      } else if (col < 2 * FEAT) {               // K
        int c = col - FEAT, h = c >> 7, d = c & 127;
        kbf[((size_t)(b * HEADS + h) * SEQ + t) * DK + d] = f2bf(val);
      } else {                                   // V (bf16 transposed + f32)
        int c = col - 2 * FEAT, h = c >> 7, d = c & 127;
        vbfT[((size_t)(b * HEADS + h) * DK + d) * SEQ + t] = f2bf(val);
        vf32[(size_t)row * FEAT + c] = val;
      }
    }
  }
}

// ---------------------------------------------------------------------------
// Kernel 2: FSMN memory: depthwise causal-centered conv(K=11) + residual, masked
// ---------------------------------------------------------------------------
__global__ void __launch_bounds__(256) fsmn_kernel(
    const float* __restrict__ vf32, const float* __restrict__ mask,
    const float* __restrict__ w, float* __restrict__ fsmn) {
  size_t idx = (size_t)blockIdx.x * 256 + threadIdx.x;
  int c = (int)(idx & 511);
  int t = (int)((idx >> 9) & 1023);
  int b = (int)(idx >> 19);
  float s = 0.f;
#pragma unroll
  for (int k = 0; k < 11; ++k) {
    int tt = t + k - 5;
    if (tt >= 0 && tt < SEQ)
      s += vf32[((size_t)b * SEQ + tt) * FEAT + c] * mask[b * SEQ + tt] * w[c * 11 + k];
  }
  float mk = mask[b * SEQ + t];
  fsmn[idx] = (s + vf32[idx] * mk) * mk;
}

// ---------------------------------------------------------------------------
// Kernel 3: flash attention per (b,h,128-query-block). 8 waves, wave w owns
// query rows [16w,16w+16). Q fragments held in registers for all 8 K-blocks.
// S = Q·Kᵀ via WMMA, online softmax (half-wave shfl reductions), P staged
// through LDS (C-layout -> A-layout), O += P·V via WMMA on transposed V.
// ---------------------------------------------------------------------------
__global__ void __launch_bounds__(256) attn_kernel(
    const unsigned short* __restrict__ qbf, const unsigned short* __restrict__ kbf,
    const unsigned short* __restrict__ vbfT, const float* __restrict__ mask,
    float* __restrict__ ctx) {
  __shared__ unsigned short sP[128 * 136];  // 34816 B; stride 136 hw = 272 B
  const int qb = blockIdx.x & 7;
  const int h  = (blockIdx.x >> 3) & 3;
  const int b  = blockIdx.x >> 5;
  const int wave = threadIdx.x >> 5;
  const int lane = threadIdx.x & 31;
  const int hl = lane >> 4, ln = lane & 15;

  const unsigned short* qp = qbf + (size_t)(b * HEADS + h) * SEQ * DK;
  const unsigned short* kp = kbf + (size_t)(b * HEADS + h) * SEQ * DK;
  const unsigned short* vp = vbfT + (size_t)(b * HEADS + h) * DK * SEQ;

  v16bf aq[4];
#pragma unroll
  for (int ks = 0; ks < 4; ++ks)
    aq[ks] = frag_at(qp, qb * 128 + wave * 16, DK, ks * 32, lane);

  const v8f vzero = {0.f, 0.f, 0.f, 0.f, 0.f, 0.f, 0.f, 0.f};
  v8f vo[8];
  float m_run[8], l_run[8];
#pragma unroll
  for (int j = 0; j < 8; ++j) vo[j] = vzero;
#pragma unroll
  for (int i = 0; i < 8; ++i) { m_run[i] = -1e30f; l_run[i] = 0.f; }

  for (int tb = 0; tb < 8; ++tb) {
    // prefetch next K/V 128-blocks toward GL2 / WGP$ (global_prefetch_b8)
    if (tb < 7) {
      __builtin_prefetch(kp + (size_t)(tb + 1) * 128 * DK + lane * 64, 0, 3);
      __builtin_prefetch(vp + (size_t)(tb + 1) * 32 + lane * 64, 0, 3);
    }
    v8f s[8];
#pragma unroll
    for (int j = 0; j < 8; ++j) s[j] = vzero;
#pragma unroll
    for (int ks = 0; ks < 4; ++ks) {
      v16bf bk[8];
#pragma unroll
      for (int j = 0; j < 8; ++j)
        bk[j] = frag_at(kp, tb * 128 + j * 16, DK, ks * 32, lane);
      sched_fence();  // 16 global b128 loads, one wait, 8 WMMAs
#pragma unroll
      for (int j = 0; j < 8; ++j) s[j] = wmma_bf16(aq[ks], bk[j], s[j]);
      sched_fence();
    }
    // key-mask: column t = tb*128 + j*16 + ln
#pragma unroll
    for (int j = 0; j < 8; ++j) {
      float mv = mask[b * SEQ + tb * 128 + j * 16 + ln];
      if (mv == 0.f) {
#pragma unroll
        for (int i = 0; i < 8; ++i) s[j][i] = -1e30f;
      }
    }
    // online softmax: rows live in 16-lane halves -> shfl_xor masks 1/2/4/8
    float nm[8];
#pragma unroll
    for (int i = 0; i < 8; ++i) nm[i] = m_run[i];
#pragma unroll
    for (int j = 0; j < 8; ++j)
#pragma unroll
      for (int i = 0; i < 8; ++i) nm[i] = fmaxf(nm[i], s[j][i]);
#pragma unroll
    for (int o = 1; o <= 8; o <<= 1)
#pragma unroll
      for (int i = 0; i < 8; ++i) nm[i] = fmaxf(nm[i], __shfl_xor(nm[i], o));

    float alpha[8], rs[8];
#pragma unroll
    for (int i = 0; i < 8; ++i) {
      alpha[i] = __expf(m_run[i] - nm[i]);
      m_run[i] = nm[i];
      rs[i] = 0.f;
    }
#pragma unroll
    for (int j = 0; j < 8; ++j) {
#pragma unroll
      for (int i = 0; i < 8; ++i) {
        float p = __expf(s[j][i] - nm[i]);
        rs[i] += p;
        sP[(size_t)(wave * 16 + i + hl * 8) * 136 + j * 16 + ln] = f2bf(p);
      }
    }
#pragma unroll
    for (int o = 1; o <= 8; o <<= 1)
#pragma unroll
      for (int i = 0; i < 8; ++i) rs[i] += __shfl_xor(rs[i], o);
#pragma unroll
    for (int i = 0; i < 8; ++i) l_run[i] = l_run[i] * alpha[i] + rs[i];
#pragma unroll
    for (int j = 0; j < 8; ++j)
#pragma unroll
      for (int i = 0; i < 8; ++i) vo[j][i] *= alpha[i];
    // O += P · V   (per-wave LDS strip; same-wave RAW handled by s_wait_dscnt)
#pragma unroll
    for (int ks = 0; ks < 4; ++ks) {
      v16bf pa = frag_at(sP, wave * 16, 136, ks * 32, lane);
      v16bf bv[8];
#pragma unroll
      for (int j = 0; j < 8; ++j)
        bv[j] = frag_at(vp, j * 16, SEQ, tb * 128 + ks * 32, lane);
      sched_fence();
#pragma unroll
      for (int j = 0; j < 8; ++j) vo[j] = wmma_bf16(pa, bv[j], vo[j]);
      sched_fence();
    }
  }

  float inv[8];
#pragma unroll
  for (int i = 0; i < 8; ++i) inv[i] = l_run[i] > 0.f ? 1.f / l_run[i] : 0.f;
#pragma unroll
  for (int j = 0; j < 8; ++j) {
#pragma unroll
    for (int i = 0; i < 8; ++i) {
      int t = qb * 128 + wave * 16 + i + hl * 8;
      int d = j * 16 + ln;
      ctx[(size_t)(b * SEQ + t) * FEAT + h * DK + d] = vo[j][i] * inv[i];
    }
  }
}

// ---------------------------------------------------------------------------
// Kernel 4: out = (ctx @ W_out + b_out + fsmn) * mask   (fused epilogue)
// ---------------------------------------------------------------------------
__global__ void __launch_bounds__(256) out_gemm_kernel(
    const float* __restrict__ ctx, const float* __restrict__ Wout,
    const float* __restrict__ bout, const float* __restrict__ fsmn,
    const float* __restrict__ mask, float* __restrict__ out) {
  __shared__ unsigned short aL[128 * 48];
  __shared__ unsigned short bL[128 * 48];
  const int m0 = blockIdx.x * 128;
  const int n0 = blockIdx.y * 128;
  const int tid = threadIdx.x;
  const int wave = tid >> 5, lane = tid & 31;
  const int hl = lane >> 4, ln = lane & 15;

  const v8f vzero = {0.f, 0.f, 0.f, 0.f, 0.f, 0.f, 0.f, 0.f};
  v8f acc[8];
#pragma unroll
  for (int j = 0; j < 8; ++j) acc[j] = vzero;

  for (int k0 = 0; k0 < FEAT; k0 += 32) {
    stage_a(aL, ctx, m0, FEAT, k0, tid);
    stage_b(bL, Wout, n0, FEAT, k0, tid);
    __syncthreads();
    v16bf af = frag_at(aL, wave * 16, 48, 0, lane);
    v16bf bfr[8];
#pragma unroll
    for (int j = 0; j < 8; ++j) bfr[j] = frag_at(bL, j * 16, 48, 0, lane);
    sched_fence();
#pragma unroll
    for (int j = 0; j < 8; ++j) acc[j] = wmma_bf16(af, bfr[j], acc[j]);
    sched_fence();
    __syncthreads();
  }

#pragma unroll
  for (int j = 0; j < 8; ++j) {
#pragma unroll
    for (int i = 0; i < 8; ++i) {
      int col = n0 + j * 16 + ln;
      int row = m0 + wave * 16 + i + hl * 8;
      size_t o = (size_t)row * FEAT + col;
      float val = acc[j][i] + bout[col] + fsmn[o];
      out[o] = val * mask[row];  // mask[b,t] flattened == mask[row]
    }
  }
}

// ---------------------------------------------------------------------------
extern "C" void kernel_launch(void* const* d_in, const int* in_sizes, int n_in,
                              void* d_out, int out_size, void* d_ws, size_t ws_size,
                              hipStream_t stream) {
  const float* x     = (const float*)d_in[0];
  const float* mask  = (const float*)d_in[1];
  const float* Wqkv  = (const float*)d_in[2];
  const float* bqkv  = (const float*)d_in[3];
  const float* Wout  = (const float*)d_in[4];
  const float* bout  = (const float*)d_in[5];
  const float* fsmnw = (const float*)d_in[6];
  float* out = (float*)d_out;

  char* ws = (char*)d_ws;
  size_t off = 0;
  const size_t bf_bytes = (size_t)BATCH * HEADS * SEQ * DK * sizeof(unsigned short);
  const size_t f_bytes  = (size_t)ROWS * FEAT * sizeof(float);
  unsigned short* qbf  = (unsigned short*)(ws + off); off += bf_bytes;  // 16 MB
  unsigned short* kbf  = (unsigned short*)(ws + off); off += bf_bytes;  // 16 MB
  unsigned short* vbfT = (unsigned short*)(ws + off); off += bf_bytes;  // 16 MB
  float* vf32 = (float*)(ws + off); off += f_bytes;                     // 32 MB
  float* fsmn = (float*)(ws + off); off += f_bytes;                     // 32 MB
  float* ctx  = (float*)(ws + off); off += f_bytes;                     // 32 MB

  qkv_gemm_kernel<<<dim3(ROWS / 128, QKVF / 128), 256, 0, stream>>>(
      x, Wqkv, bqkv, qbf, kbf, vbfT, vf32);
  fsmn_kernel<<<dim3((ROWS * FEAT) / 256), 256, 0, stream>>>(vf32, mask, fsmnw, fsmn);
  attn_kernel<<<dim3(BATCH * HEADS * (SEQ / 128)), 256, 0, stream>>>(
      qbf, kbf, vbfT, mask, ctx);
  out_gemm_kernel<<<dim3(ROWS / 128, FEAT / 128), 256, 0, stream>>>(
      ctx, Wout, bout, fsmn, mask, out);
}
